// NeighborhoodAttention_6622839571046
// MI455X (gfx1250) — matrix operand order repaired and checked by reading, not compiled, not run on hardware
//
#include <hip/hip_runtime.h>
#include <hip/hip_bf16.h>
#include <stdint.h>

// Problem constants (match reference)
#define EMBED   512
#define NH      8
#define HD      64
#define T_      8
#define H_      32
#define W_      32
#define B_      2
#define NTOK    (B_*T_*H_*W_)   // 16384
#define QKVN    (3*EMBED)       // 1536
#define NKEY    72              // t*3*3
#define NKEYP   80              // padded to 5 tiles of 16
#define NKEYPP  96              // padded to 3 K-tiles of 32

typedef __attribute__((ext_vector_type(16))) __bf16 v16bf;
typedef __attribute__((ext_vector_type(8)))  float  v8f;

union Frag { uint32_t u[8]; v16bf v; };

__device__ __forceinline__ unsigned short f2bf(float f) {
  union { float f; uint32_t u; } a; a.f = f;
  uint32_t r = a.u + 0x7FFFu + ((a.u >> 16) & 1u);   // round-to-nearest-even
  return (unsigned short)(r >> 16);
}
__device__ __forceinline__ float bf2f(unsigned short h) {
  union { uint32_t u; float f; } a; a.u = ((uint32_t)h) << 16;
  return a.f;
}
// A-matrix 16x32 bf16 per-lane K offsets (contiguous, matching B frags): VGPR j holds K pair 2j + 16*hf
__device__ __forceinline__ int kOffA(int j, int hf) {
  return 2*j + 16*hf;
}

// ---------------------------------------------------------------- convert ---
__global__ void k_f32_to_bf16(const float* __restrict__ in,
                              unsigned short* __restrict__ out, int n) {
  int i = blockIdx.x * blockDim.x + threadIdx.x;
  if (i < n) out[i] = f2bf(in[i]);
}

// ------------------------------------------------------------------- GEMM ---
// C[M,N] = A[M,K] * Bw[N,K]^T  (both bf16, f32 accumulate)
// One wave computes a 32(M) x 64(N) tile: 2 A-frags + 4 B-frags -> 8 chained
// WMMAs per 32-K step (8 wmma : 12 b128 loads).
template <bool STORE_BF16>
__global__ void __launch_bounds__(32)
k_gemm(const unsigned short* __restrict__ A,
       const unsigned short* __restrict__ Bw,
       unsigned short* __restrict__ Cb,
       float* __restrict__ Cf,
       int M, int N, int K) {
  const int lane = threadIdx.x & 31;
  const int l = lane & 15, hf = lane >> 4;
  const int nBase = blockIdx.x * 64;
  const int mBase = blockIdx.y * 32;

  const uint32_t* Ar0 = (const uint32_t*)(A  + (size_t)(mBase +      l) * K);
  const uint32_t* Ar1 = (const uint32_t*)(A  + (size_t)(mBase + 16 + l) * K);
  const uint32_t* Br0 = (const uint32_t*)(Bw + (size_t)(nBase +  0 + l) * K);
  const uint32_t* Br1 = (const uint32_t*)(Bw + (size_t)(nBase + 16 + l) * K);
  const uint32_t* Br2 = (const uint32_t*)(Bw + (size_t)(nBase + 32 + l) * K);
  const uint32_t* Br3 = (const uint32_t*)(Bw + (size_t)(nBase + 48 + l) * K);

  v8f a00 = {}, a01 = {}, a02 = {}, a03 = {};   // M rows 0..15 of tile
  v8f a10 = {}, a11 = {}, a12 = {}, a13 = {};   // M rows 16..31 of tile

  for (int kb = 0; kb < K; kb += 32) {
    Frag fa0, fa1, b0, b1, b2, b3;
    const int k2 = kb >> 1;
#pragma unroll
    for (int j = 0; j < 8; ++j) {
      const int o = (kb + kOffA(j, hf)) >> 1;
      fa0.u[j] = Ar0[o];
      fa1.u[j] = Ar1[o];
    }
#pragma unroll
    for (int j = 0; j < 8; ++j) {
      const int o = k2 + j + 8*hf;      // B frag: lane=N col, K = {2j,2j+1}(+16*hf)
      b0.u[j] = Br0[o]; b1.u[j] = Br1[o]; b2.u[j] = Br2[o]; b3.u[j] = Br3[o];
    }
    a00 = __builtin_amdgcn_wmma_f32_16x16x32_bf16(false, fa0.v, false, b0.v, (short)0, a00, false, false);
    a01 = __builtin_amdgcn_wmma_f32_16x16x32_bf16(false, fa0.v, false, b1.v, (short)0, a01, false, false);
    a02 = __builtin_amdgcn_wmma_f32_16x16x32_bf16(false, fa0.v, false, b2.v, (short)0, a02, false, false);
    a03 = __builtin_amdgcn_wmma_f32_16x16x32_bf16(false, fa0.v, false, b3.v, (short)0, a03, false, false);
    a10 = __builtin_amdgcn_wmma_f32_16x16x32_bf16(false, fa1.v, false, b0.v, (short)0, a10, false, false);
    a11 = __builtin_amdgcn_wmma_f32_16x16x32_bf16(false, fa1.v, false, b1.v, (short)0, a11, false, false);
    a12 = __builtin_amdgcn_wmma_f32_16x16x32_bf16(false, fa1.v, false, b2.v, (short)0, a12, false, false);
    a13 = __builtin_amdgcn_wmma_f32_16x16x32_bf16(false, fa1.v, false, b3.v, (short)0, a13, false, false);
  }

  // C layout: reg r -> row r (+8 for upper half-wave), lane -> col
#pragma unroll
  for (int r = 0; r < 8; ++r) {
    const size_t row0 = (size_t)(mBase +      8*hf + r);
    const size_t row1 = (size_t)(mBase + 16 + 8*hf + r);
    if (STORE_BF16) {
      Cb[row0*N + nBase +  0 + l] = f2bf(a00[r]);
      Cb[row0*N + nBase + 16 + l] = f2bf(a01[r]);
      Cb[row0*N + nBase + 32 + l] = f2bf(a02[r]);
      Cb[row0*N + nBase + 48 + l] = f2bf(a03[r]);
      Cb[row1*N + nBase +  0 + l] = f2bf(a10[r]);
      Cb[row1*N + nBase + 16 + l] = f2bf(a11[r]);
      Cb[row1*N + nBase + 32 + l] = f2bf(a12[r]);
      Cb[row1*N + nBase + 48 + l] = f2bf(a13[r]);
    } else {
      Cf[row0*N + nBase +  0 + l] = a00[r];
      Cf[row0*N + nBase + 16 + l] = a01[r];
      Cf[row0*N + nBase + 32 + l] = a02[r];
      Cf[row0*N + nBase + 48 + l] = a03[r];
      Cf[row1*N + nBase +  0 + l] = a10[r];
      Cf[row1*N + nBase + 16 + l] = a11[r];
      Cf[row1*N + nBase + 32 + l] = a12[r];
      Cf[row1*N + nBase + 48 + l] = a13[r];
    }
  }
}

// --------------------------------------------------- RMSNorm + rotary (q,k) ---
// One thread handles one (token, head, q|k) row of 64 bf16 values, in place.
__global__ void k_rmsrope(unsigned short* __restrict__ qkv,
                          const float* __restrict__ freqs,
                          const float* __restrict__ qg,
                          const float* __restrict__ kg) {
  int idx = blockIdx.x * blockDim.x + threadIdx.x;   // token*16 + g*2 + isK
  if (idx >= NTOK * NH * 2) return;
  const int token = idx >> 4;
  const int g     = (idx >> 1) & 7;
  const int isK   = idx & 1;
  unsigned short* p = qkv + (size_t)token * QKVN + g * 192 + isK * 64;
  const float* gamma = isK ? kg : qg;
  const float* fr = freqs + (size_t)(token & (T_*H_*W_ - 1)) * (HD/2);

  float x[64];
  float ss = 0.f;
#pragma unroll
  for (int i = 0; i < 64; ++i) { x[i] = bf2f(p[i]); ss += x[i]*x[i]; }
  const float inv = rsqrtf(ss * (1.0f/64.0f) + 1e-6f);
#pragma unroll
  for (int i = 0; i < 32; ++i) {
    const float xe = x[2*i]   * inv * gamma[2*i];
    const float xo = x[2*i+1] * inv * gamma[2*i+1];
    float s, c; __sincosf(fr[i], &s, &c);
    p[2*i]   = f2bf(xe*c - xo*s);
    p[2*i+1] = f2bf(xe*s + xo*c);
  }
}

// ------------------------------------------------------ neighborhood attn ---
// One wave per (b,h,w,head): Q=8x64 (padded to 16 rows), 72 keys (pad 80/96).
__global__ void __launch_bounds__(32)
k_attn(const unsigned short* __restrict__ qkv,
       unsigned short* __restrict__ out) {
  __shared__ unsigned short s_kv[64 * 96];  // K stage [80][64] then V^T [64][96]
  __shared__ unsigned short s_p [16 * 96];  // P (softmax probs) [16][96]

  const int lane = threadIdx.x;
  const int l = lane & 15, hf = lane >> 4;
  const int bid = blockIdx.x;
  const int g = bid & 7, w = (bid >> 3) & 31, h = (bid >> 8) & 31, b = bid >> 13;
  const int h0 = min(max(h - 1, 0), H_ - 3);
  const int w0 = min(max(w - 1, 0), W_ - 3);

  uint32_t* kv32 = (uint32_t*)s_kv;
  uint32_t* p32  = (uint32_t*)s_p;
  const uint32_t* q32 = (const uint32_t*)qkv;   // row stride 1536 u16 = 768 u32

  // ---- stage K window into LDS [key][64] via CDNA5 async DMA (no VGPR
  //      round-trip, tracked with ASYNCcnt); zero-pad keys 72..79 with ds.
  for (int e = lane; e < NKEYP * 8; e += 32) {   // 8 x 16B chunks per key row
    const int key = e >> 3, ch = e & 7;
    if (key < NKEY) {
      const int t = key / 9, r = key % 9;
      const int tok = ((b*T_ + t)*H_ + (h0 + r/3))*W_ + (w0 + r%3);
      const unsigned short* src = qkv + (size_t)tok*QKVN + g*192 + 64 + ch*8;
      asm volatile("global_load_async_to_lds_b128 %0, %1, off"
                   :: "v"((uint32_t)(uintptr_t)(s_kv + key*64 + ch*8)),
                      "v"((uint64_t)(uintptr_t)src)
                   : "memory");
    } else {
      uint32_t* dst = kv32 + key*32 + ch*4;
      dst[0] = 0; dst[1] = 0; dst[2] = 0; dst[3] = 0;
    }
  }
  asm volatile("s_wait_asynccnt 0x0" ::: "memory");
  __syncthreads();

  // ---- Q fragments (rows 8..15 zero)
  Frag aQ0, aQ1;
  if (l < T_) {
    const int tok = ((b*T_ + l)*H_ + h)*W_ + w;
    const uint32_t* Qr = q32 + (size_t)tok * 768 + g*96;
#pragma unroll
    for (int j = 0; j < 8; ++j) {
      aQ0.u[j] = Qr[ kOffA(j, hf)       >> 1];
      aQ1.u[j] = Qr[(kOffA(j, hf) + 32) >> 1];
    }
  } else {
#pragma unroll
    for (int j = 0; j < 8; ++j) { aQ0.u[j] = 0; aQ1.u[j] = 0; }
  }

  // ---- scores: 5 key tiles x (K=64 via 2 chained WMMAs)
  v8f sc[5];
#pragma unroll
  for (int nt = 0; nt < 5; ++nt) {
    Frag b0, b1;
    const int key = nt*16 + l;
#pragma unroll
    for (int j = 0; j < 8; ++j) {
      b0.u[j] = kv32[key*32 +      j + 8*hf];
      b1.u[j] = kv32[key*32 + 16 + j + 8*hf];
    }
    v8f s = {};
    s = __builtin_amdgcn_wmma_f32_16x16x32_bf16(false, aQ0.v, false, b0.v, (short)0, s, false, false);
    s = __builtin_amdgcn_wmma_f32_16x16x32_bf16(false, aQ1.v, false, b1.v, (short)0, s, false, false);
    sc[nt] = s;
  }
  __syncthreads();   // K-window reads done before V overwrites s_kv

  // ---- scale + mask (keys >= 72) + softmax across 16 lanes (half-wave)
  const float scale = 0.125f;   // 64^-0.5
  float p[5][8];
#pragma unroll
  for (int r = 0; r < 8; ++r) {
    const float s0 = sc[0][r]*scale, s1 = sc[1][r]*scale;
    const float s2 = sc[2][r]*scale, s3 = sc[3][r]*scale;
    const float s4 = (l < 8) ? sc[4][r]*scale : -1e30f;
    float mx = fmaxf(fmaxf(fmaxf(s0, s1), fmaxf(s2, s3)), s4);
#pragma unroll
    for (int d = 1; d < 16; d <<= 1) mx = fmaxf(mx, __shfl_xor(mx, d, 32));
    const float e0 = __expf(s0 - mx), e1 = __expf(s1 - mx);
    const float e2 = __expf(s2 - mx), e3 = __expf(s3 - mx);
    const float e4 = (l < 8) ? __expf(s4 - mx) : 0.f;
    float sum = e0 + e1 + e2 + e3 + e4;
#pragma unroll
    for (int d = 1; d < 16; d <<= 1) sum += __shfl_xor(sum, d, 32);
    const float rs = 1.0f / sum;
    p[0][r] = e0*rs; p[1][r] = e1*rs; p[2][r] = e2*rs; p[3][r] = e3*rs; p[4][r] = e4*rs;
  }

  // ---- zero P and V^T regions, stage V^T [hd][key stride 96], store P
  for (int e = lane; e < (16*96)/2; e += 32) p32[e] = 0;
  for (int e = lane; e < (64*96)/2; e += 32) kv32[e] = 0;
  __syncthreads();
  for (int e = lane; e < NKEY * 32; e += 32) {
    const int key = e >> 5, dw = e & 31;
    const int t = key / 9, r = key % 9;
    const int tok = ((b*T_ + t)*H_ + (h0 + r/3))*W_ + (w0 + r%3);
    const uint32_t v = q32[(size_t)tok * 768 + g*96 + 64 + dw];   // v part
    s_kv[(2*dw    )*96 + key] = (unsigned short)(v & 0xFFFFu);
    s_kv[(2*dw + 1)*96 + key] = (unsigned short)(v >> 16);
  }
  if (hf == 0) {   // only real query rows 0..7
#pragma unroll
    for (int nt = 0; nt < 5; ++nt)
#pragma unroll
      for (int r = 0; r < 8; ++r)
        s_p[r*96 + nt*16 + l] = f2bf(p[nt][r]);
  }
  __syncthreads();

  // ---- out = P (16x96) x V (96x64): 4 N-tiles x 3 chained WMMAs
  Frag aP[3];
#pragma unroll
  for (int kb = 0; kb < 3; ++kb)
#pragma unroll
    for (int j = 0; j < 8; ++j)
      aP[kb].u[j] = p32[l*48 + ((kb*32 + kOffA(j, hf)) >> 1)];

  const int tokBase = ((b*T_)*H_ + h)*W_ + w;
#pragma unroll
  for (int nt = 0; nt < 4; ++nt) {
    v8f o = {};
    const int n = nt*16 + l;
#pragma unroll
    for (int kb = 0; kb < 3; ++kb) {
      Frag bv;
#pragma unroll
      for (int j = 0; j < 8; ++j)
        bv.u[j] = kv32[n*48 + kb*16 + j + 8*hf];   // V^T: lane=hd col, K=key pairs
      o = __builtin_amdgcn_wmma_f32_16x16x32_bf16(false, aP[kb].v, false, bv.v, (short)0, o, false, false);
    }
    if (hf == 0) {
#pragma unroll
      for (int r = 0; r < 8; ++r) {
        const int tok = tokBase + r * (H_*W_);     // query time step r
        out[(size_t)tok * EMBED + g*64 + nt*16 + l] = f2bf(o[r]);
      }
    }
  }
}

// ---------------------------------------------------------------- launch ---
extern "C" void kernel_launch(void* const* d_in, const int* in_sizes, int n_in,
                              void* d_out, int out_size, void* d_ws, size_t ws_size,
                              hipStream_t stream) {
  const float* x     = (const float*)d_in[0];   // (2,8,32,32,512)
  const float* freqs = (const float*)d_in[1];   // (8,32,32,32)
  const float* w_in  = (const float*)d_in[2];   // (1536,512)
  const float* w_out = (const float*)d_in[3];   // (512,512)
  const float* qg    = (const float*)d_in[4];   // (64,)
  const float* kg    = (const float*)d_in[5];   // (64,)

  char* ws = (char*)d_ws;
  const size_t nX   = (size_t)NTOK * EMBED;     // 8388608
  const size_t nWin = (size_t)QKVN * EMBED;     // 786432
  const size_t nWo  = (size_t)EMBED * EMBED;    // 262144
  unsigned short* x_bf    = (unsigned short*)(ws);
  unsigned short* win_bf  = (unsigned short*)(ws + nX*2);
  unsigned short* wout_bf = (unsigned short*)(ws + nX*2 + nWin*2);
  unsigned short* qkv_bf  = (unsigned short*)(ws + nX*2 + nWin*2 + nWo*2);
  unsigned short* ao_bf   = (unsigned short*)(ws + nX*2 + nWin*2 + nWo*2 + (size_t)NTOK*QKVN*2);

  k_f32_to_bf16<<<(int)((nX   + 255)/256), 256, 0, stream>>>(x,     x_bf,    (int)nX);
  k_f32_to_bf16<<<(int)((nWin + 255)/256), 256, 0, stream>>>(w_in,  win_bf,  (int)nWin);
  k_f32_to_bf16<<<(int)((nWo  + 255)/256), 256, 0, stream>>>(w_out, wout_bf, (int)nWo);

  // QKV projection: 16384 x 1536 x 512
  k_gemm<true><<<dim3(QKVN/64, NTOK/32), 32, 0, stream>>>(
      x_bf, win_bf, qkv_bf, nullptr, NTOK, QKVN, EMBED);

  // RMSNorm + rotary on q,k in place
  k_rmsrope<<<(NTOK*NH*2)/256, 256, 0, stream>>>(qkv_bf, freqs, qg, kg);

  // Neighborhood attention: one wave per (b,h,w,head)
  k_attn<<<B_*H_*W_*NH, 32, 0, stream>>>(qkv_bf, ao_bf);

  // Output projection: 16384 x 512 x 512 -> f32 d_out
  k_gemm<false><<<dim3(EMBED/64, NTOK/32), 32, 0, stream>>>(
      ao_bf, wout_bf, nullptr, (float*)d_out, NTOK, EMBED, EMBED);
}
